// GraphAttention_49847390437980
// MI455X (gfx1250) — matrix operand order repaired and checked by reading, not compiled, not run on hardware
//
#include <hip/hip_runtime.h>

// Problem constants (match reference)
#define CDIM 128
#define NDIM 2048
#define BDIM 8
#define NCOLS 144   // 128 output cols + col 128 = denominator + 15 pad (discarded)

typedef __attribute__((ext_vector_type(16))) _Float16 v16h;
typedef __attribute__((ext_vector_type(8)))  float    v8f;

// ---------------------------------------------------------------------------
// K1: v[c] = sum_d W_w[d,c]*a_w[C+d] ; c0 = dot(W_b, a_w[C:]).  (si & a_b cancel
// in the row-softmax, so the Linear(C,C) GEMM collapses to one 128-vector.)
// ---------------------------------------------------------------------------
__global__ void ga_prep(const float* __restrict__ Ww, const float* __restrict__ Wb,
                        const float* __restrict__ aw,
                        float* __restrict__ v, float* __restrict__ c0) {
  int c = threadIdx.x;                       // 128 threads
  const float* a2 = aw + CDIM;
  float s = 0.f;
  #pragma unroll 4
  for (int d = 0; d < CDIM; ++d) s = fmaf(Ww[d * CDIM + c], a2[d], s);
  v[c] = s;
  __shared__ float red[CDIM];
  red[c] = Wb[c] * a2[c];
  __syncthreads();
  for (int off = CDIM / 2; off > 0; off >>= 1) {
    if (c < off) red[c] += red[c + off];
    __syncthreads();
  }
  if (c == 0) *c0 = red[0];
}

// ---------------------------------------------------------------------------
// K2a: w[b,n] = exp(h[b,n,:].v + c0)   (one wave32 per row)
// ---------------------------------------------------------------------------
__global__ void ga_scores(const float* __restrict__ h, const float* __restrict__ v,
                          const float* __restrict__ c0, float* __restrict__ w) {
  int wv = threadIdx.x >> 5, lane = threadIdx.x & 31;
  long row = (long)blockIdx.x * 8 + wv;      // 8 waves/block, row in [0, B*N)
  const float* hr = h + row * CDIM;
  float s = 0.f;
  #pragma unroll
  for (int k = lane; k < CDIM; k += 32) s = fmaf(hr[k], v[k], s);
  #pragma unroll
  for (int off = 16; off > 0; off >>= 1) s += __shfl_xor(s, off, 32);
  if (lane == 0) w[row] = expf(s + *c0);
}

// ---------------------------------------------------------------------------
// K2b: HwT[b, c, n] = f16( w[b,n] * h[b,n,c] )   (32x32 LDS transpose, coalesced)
// ---------------------------------------------------------------------------
__global__ void ga_hwT(const float* __restrict__ h, const float* __restrict__ w,
                       _Float16* __restrict__ HwT) {
  __shared__ float tile[32][33];
  __shared__ float wsh[32];
  const int b  = blockIdx.z;
  const int n0 = blockIdx.y * 32, c0 = blockIdx.x * 32;
  const int tx = threadIdx.x, ty = threadIdx.y;            // (32, 8)
  #pragma unroll
  for (int j = 0; j < 32; j += 8)
    tile[ty + j][tx] = h[((long)b * NDIM + (n0 + ty + j)) * CDIM + c0 + tx];
  if (ty == 0) wsh[tx] = w[(long)b * NDIM + n0 + tx];
  __syncthreads();
  #pragma unroll
  for (int j = 0; j < 32; j += 8)
    HwT[((long)b * NCOLS + (c0 + ty + j)) * NDIM + n0 + tx] =
        (_Float16)(tile[tx][ty + j] * wsh[tx]);
}

// K2c: row 128 of HwT = w (denominator column); rows 129..143 = 0 (pad)
__global__ void ga_hwT_pad(const float* __restrict__ w, _Float16* __restrict__ HwT) {
  const int b = blockIdx.y;
  const int n = blockIdx.x * 256 + threadIdx.x;
  HwT[((long)b * NCOLS + 128) * NDIM + n] = (_Float16)w[(long)b * NDIM + n];
  #pragma unroll
  for (int r = 129; r < NCOLS; ++r)
    HwT[((long)b * NCOLS + r) * NDIM + n] = (_Float16)0.f;
}

// ---------------------------------------------------------------------------
// K3: out[b,i,:] = (A[b,i,:] @ (w.h)[b]) / (A[b,i,:] @ w[b])
// One wave per 16 output rows; 9 n-tiles (8 data + 1 denom); K-loop of 32.
// All 22 b128 loads of an iteration are issued before the 9-WMMA sequence so
// the scheduler can keep them in flight (low occupancy -> ILP must hide L2).
// ---------------------------------------------------------------------------
__global__ void __launch_bounds__(128)
ga_gemm(const float* __restrict__ A, const _Float16* __restrict__ HwT,
        float* __restrict__ out) {
  const int lane = threadIdx.x & 31;
  const int wv   = threadIdx.x >> 5;         // 4 waves/block
  const int g    = lane >> 4;                // half-wave group (fragment layout)
  const int ml   = lane & 15;
  const int b    = blockIdx.y;
  const int i0   = blockIdx.x * 64 + wv * 16;

  const float*    Arow = A   + ((size_t)b * NDIM + (i0 + ml)) * NDIM;
  const _Float16* Hb   = HwT + (size_t)b * NCOLS * NDIM + (size_t)ml * NDIM;

  v8f acc[9] = {};

  for (int j0 = 0; j0 < NDIM; j0 += 32) {
    // ---- issue A raw loads first (1 contiguous run of 16 f32: k = j0 + g*16 + 0..15)
    const float* ap = Arow + j0 + g * 16;
    float4 r0 = *(const float4*)(ap);
    float4 r1 = *(const float4*)(ap + 4);
    float4 r2 = *(const float4*)(ap + 8);
    float4 r3 = *(const float4*)(ap + 12);

    // ---- issue ALL 9 B-fragment loads (32B contiguous per lane, L2-resident)
    v16h bf[9];
    #pragma unroll
    for (int t = 0; t < 9; ++t)
      bf[t] = *(const v16h*)(Hb + (size_t)t * 16 * NDIM + j0 + g * 16);

    if (j0 + 32 < NDIM) __builtin_prefetch(ap + 32, 0, 3);   // global_prefetch_b8

    // ---- convert A to f16 fragment while B loads are in flight
    //   elems 0..15 -> K = j0 + g*16 + {0..15}
    v16h af;
    af[0]  = (_Float16)r0.x; af[1]  = (_Float16)r0.y; af[2]  = (_Float16)r0.z; af[3]  = (_Float16)r0.w;
    af[4]  = (_Float16)r1.x; af[5]  = (_Float16)r1.y; af[6]  = (_Float16)r1.z; af[7]  = (_Float16)r1.w;
    af[8]  = (_Float16)r2.x; af[9]  = (_Float16)r2.y; af[10] = (_Float16)r2.z; af[11] = (_Float16)r2.w;
    af[12] = (_Float16)r3.x; af[13] = (_Float16)r3.y; af[14] = (_Float16)r3.z; af[15] = (_Float16)r3.w;

    // ---- 9 WMMAs; waits on bf[t] can now be graduated instead of full drains
    #pragma unroll
    for (int t = 0; t < 9; ++t)
      acc[t] = __builtin_amdgcn_wmma_f32_16x16x32_f16(
          /*neg_a=*/false, af, /*neg_b=*/false, bf[t],
          /*c_mod=*/(short)0, acc[t], /*reuse_a=*/false, /*reuse_b=*/false);
  }

  // ---- denominator broadcast: acc[8] col n=0 lives in lanes 0 (m=0..7) / 16 (m=8..15)
  float inv[8];
  #pragma unroll
  for (int r = 0; r < 8; ++r) {
    float den = __shfl(acc[8][r], g << 4, 32);   // den[m = r + g*8]
    inv[r] = 1.0f / den;
  }

  // ---- D layout: lane holds (m = r + g*8, n = ml); store f32 normalized output
  float* orow = out + ((size_t)b * NDIM + i0 + g * 8) * CDIM + ml;
  #pragma unroll
  for (int r = 0; r < 8; ++r) {
    #pragma unroll
    for (int t = 0; t < 8; ++t)
      orow[(size_t)r * CDIM + t * 16] = acc[t][r] * inv[r];
  }
}

// ---------------------------------------------------------------------------
extern "C" void kernel_launch(void* const* d_in, const int* in_sizes, int n_in,
                              void* d_out, int out_size, void* d_ws, size_t ws_size,
                              hipStream_t stream) {
  const float* h  = (const float*)d_in[0];   // [B,N,C]
  const float* A  = (const float*)d_in[1];   // [B,N,N]
  const float* Ww = (const float*)d_in[2];   // [C,C]
  const float* Wb = (const float*)d_in[3];   // [C]
  const float* aw = (const float*)d_in[4];   // [2C]
  // d_in[5] = a_b : cancels in softmax, unused
  float* out = (float*)d_out;                // [B,N,C] f32

  char* ws = (char*)d_ws;
  float*    v   = (float*)ws;                                  // 128 f32
  float*    c0  = (float*)(ws + 512);                          // 1 f32
  float*    w   = (float*)(ws + 1024);                         // B*N f32
  _Float16* HwT = (_Float16*)(ws + 1024 + sizeof(float) * BDIM * NDIM); // [B,144,N] f16

  ga_prep  <<<1, 128, 0, stream>>>(Ww, Wb, aw, v, c0);
  ga_scores<<<(BDIM * NDIM) / 8, 256, 0, stream>>>(h, v, c0, w);
  ga_hwT   <<<dim3(CDIM / 32, NDIM / 32, BDIM), dim3(32, 8), 0, stream>>>(h, w, HwT);
  ga_hwT_pad<<<dim3(NDIM / 256, BDIM), 256, 0, stream>>>(w, HwT);
  ga_gemm  <<<dim3(NDIM / 64, BDIM), 128, 0, stream>>>(A, HwT, out);
}
